// GCNModelAE_31224412242680
// MI455X (gfx1250) — matrix operand order repaired and checked by reading, not compiled, not run on hardware
//
#include <hip/hip_runtime.h>
#include <hip/hip_bf16.h>

// ---------------------------------------------------------------------------
// GCN autoencoder forward on gfx1250 (MI455X, wave32, WMMA).
//   h   = relu(spmm(x @ W1));  z = spmm(h @ W2);  recon = z @ z^T
// All GEMMs via v_wmma_f32_16x16x32_bf16 (fp32 accumulate).
// SpMM via CSR built from sorted edge_row (binary search), one row per block,
// register accumulation -> no atomics, deterministic.
// Decoder GEMM uses non-temporal C stores (268 MB write-once stream past L2).
// ---------------------------------------------------------------------------

typedef __attribute__((ext_vector_type(16))) __bf16 v16bf;
typedef __attribute__((ext_vector_type(8)))  __bf16 v8bf;
typedef __attribute__((ext_vector_type(8)))  float  v8f;

#define N_NODES 8192
#define N_EDGES 262144
#define IN_DIM  512
#define HID_DIM 256
#define Z_DIM   64

__device__ __forceinline__ unsigned short f32_to_bf16_bits(float x) {
    unsigned int u = __float_as_uint(x);
    u += 0x7FFFu + ((u >> 16) & 1u);          // round-to-nearest-even
    return (unsigned short)(u >> 16);
}

// ---------------- elementwise f32 -> bf16 ----------------
__global__ __launch_bounds__(256) void cvt_f32_bf16(const float* __restrict__ src,
                                                    unsigned short* __restrict__ dst,
                                                    int n) {
    int i = blockIdx.x * 256 + threadIdx.x;
    if (i < n) dst[i] = f32_to_bf16_bits(src[i]);
}

// -------- transpose + convert: src[R][C] f32 -> dst[C][R] bf16 --------
__global__ __launch_bounds__(256) void transpose_cvt_bf16(const float* __restrict__ src,
                                                          unsigned short* __restrict__ dst,
                                                          int R, int C) {
    int i = blockIdx.x * 256 + threadIdx.x;
    if (i >= R * C) return;
    int r = i / C, c = i - r * C;
    dst[c * R + r] = f32_to_bf16_bits(src[i]);
}

// -------- CSR row pointers from sorted edge_row (lower_bound) --------
__global__ __launch_bounds__(256) void build_row_ptr(const int* __restrict__ edge_row,
                                                     int* __restrict__ row_ptr,
                                                     int E, int nrows) {
    int i = blockIdx.x * 256 + threadIdx.x;
    if (i > nrows) return;
    int lo = 0, hi = E;
    while (lo < hi) {
        int mid = (lo + hi) >> 1;
        if (edge_row[mid] < i) lo = mid + 1; else hi = mid;
    }
    row_ptr[i] = lo;
}

// ---------------------------------------------------------------------------
// NT GEMM: C[M][ldc] (f32) = A[M][K] (bf16, row-major) . Bt[N][K]^T (bf16)
// Per wave: 16(M) x 64(N) tile = 4 x v_wmma_f32_16x16x32_bf16 accumulators.
// 8 waves per block -> 128 x 64 output per workgroup.
//
// Fragment layouts as loaded here (16-bit, wave32; contiguous K, same order for A and B):
//   A 16x32 : lanes 0-15 row M=lane, K 0..15; lanes 16-31 same row,
//             K 16..31  -> two contiguous 16B loads per lane.
//   B 32x16 : lanes 0-15 col N=lane, K 0..15; lanes 16-31 K 16..31
//             -> one contiguous 32B run per lane (two 16B loads).
//   C 16x16 : VGPR r -> row m0+r (lanes 0-15) / m0+8+r (lanes 16-31), N=lane%16.
//
// Inner loop materializes A + all four B fragments BEFORE any WMMA so the
// backend can clause the 10 b128 loads behind one s_wait and then issue the
// 4 WMMAs back-to-back (distinct accumulators -> no D->A/B hazards).
// ---------------------------------------------------------------------------
template <bool STREAM_C>
__global__ __launch_bounds__(256) void gemm_nt_bf16_wmma(
    const unsigned short* __restrict__ A,
    const unsigned short* __restrict__ Bt,
    float* __restrict__ C,
    int M, int N, int K, int ldc)
{
    const int lane = threadIdx.x & 31;
    const int wave = threadIdx.x >> 5;
    const int m0   = (blockIdx.y * 8 + wave) * 16;
    const int n0   = blockIdx.x * 64;
    if (m0 >= M || n0 >= N) return;           // uniform per wave: EXEC stays all-1s

    const int half = lane >> 4;               // 0 = lanes 0-15, 1 = lanes 16-31
    const int l16  = lane & 15;

    const unsigned short* arow = A + (size_t)(m0 + l16) * K + half * 16;
    const unsigned short* brow0 = Bt + (size_t)(n0 +  0 + l16) * K + half * 16;
    const unsigned short* brow1 = Bt + (size_t)(n0 + 16 + l16) * K + half * 16;
    const unsigned short* brow2 = Bt + (size_t)(n0 + 32 + l16) * K + half * 16;
    const unsigned short* brow3 = Bt + (size_t)(n0 + 48 + l16) * K + half * 16;

    v8f acc[4] = {v8f{}, v8f{}, v8f{}, v8f{}};

    for (int k = 0; k < K; k += 32) {
        union { v16bf v; v8bf h[2]; } afrag, bfrag[4];
        // Issue every load for this K-step first (one clause, one wait)...
        afrag.h[0]    = *(const v8bf*)(arow  + k);
        afrag.h[1]    = *(const v8bf*)(arow  + k + 8);
        bfrag[0].h[0] = *(const v8bf*)(brow0 + k);
        bfrag[0].h[1] = *(const v8bf*)(brow0 + k + 8);
        bfrag[1].h[0] = *(const v8bf*)(brow1 + k);
        bfrag[1].h[1] = *(const v8bf*)(brow1 + k + 8);
        bfrag[2].h[0] = *(const v8bf*)(brow2 + k);
        bfrag[2].h[1] = *(const v8bf*)(brow2 + k + 8);
        bfrag[3].h[0] = *(const v8bf*)(brow3 + k);
        bfrag[3].h[1] = *(const v8bf*)(brow3 + k + 8);
        // ...then 4 back-to-back WMMAs on independent accumulators.
#pragma unroll
        for (int t = 0; t < 4; ++t)
            acc[t] = __builtin_amdgcn_wmma_f32_16x16x32_bf16(
                false, afrag.v, false, bfrag[t].v, (short)0, acc[t], false, false);
    }

#pragma unroll
    for (int t = 0; t < 4; ++t) {
        float* cptr = C + (size_t)(m0 + half * 8) * ldc + (n0 + t * 16 + l16);
#pragma unroll
        for (int r = 0; r < 8; ++r) {
            if (STREAM_C)
                __builtin_nontemporal_store(acc[t][r], cptr + (size_t)r * ldc);
            else
                cptr[(size_t)r * ldc] = acc[t][r];
        }
    }
}

// ---------------------------------------------------------------------------
// SpMM: out[row][d] = sum_{e in row} w[e] * H[col[e]][d]   (+ optional relu)
// One block per row, D threads. H is L2-resident (<= 8 MB), row reads are
// fully coalesced (D consecutive floats per edge). No atomics.
// ---------------------------------------------------------------------------
template <int D, bool RELU, bool WRITE_F32>
__global__ void spmm_kernel(const float* __restrict__ H,
                            const int* __restrict__ ecol,
                            const float* __restrict__ ew,
                            const int* __restrict__ rowptr,
                            float* __restrict__ outF,
                            unsigned short* __restrict__ outB)
{
    const int row = blockIdx.x;
    const int d   = threadIdx.x;
    const int beg = rowptr[row];
    const int end = rowptr[row + 1];

    float acc = 0.0f;
    for (int e = beg; e < end; ++e) {
        const int   c = ecol[e];
        const float w = ew[e];
        acc = fmaf(w, H[(size_t)c * D + d], acc);
    }
    if (RELU) acc = fmaxf(acc, 0.0f);

    const size_t o = (size_t)row * D + d;
    if (WRITE_F32) outF[o] = acc;
    outB[o] = f32_to_bf16_bits(acc);
}

// ---------------------------------------------------------------------------
extern "C" void kernel_launch(void* const* d_in, const int* in_sizes, int n_in,
                              void* d_out, int out_size, void* d_ws, size_t ws_size,
                              hipStream_t stream) {
    const float* x    = (const float*)d_in[0];   // [8192][512]
    const float* w1   = (const float*)d_in[1];   // [512][256]
    const float* w2   = (const float*)d_in[2];   // [256][64]
    const int*   erow = (const int*)  d_in[3];   // [E] sorted
    const int*   ecol = (const int*)  d_in[4];   // [E]
    const float* ew   = (const float*)d_in[5];   // [E]

    float* recon = (float*)d_out;                              // [8192][8192]
    float* z_out = (float*)d_out + (size_t)N_NODES * N_NODES;  // [8192][64]

    // ---- workspace layout (all 16B-aligned offsets) ----
    char* ws = (char*)d_ws;
    unsigned short* x_bf  = (unsigned short*)(ws + 0);          //  8 MB
    unsigned short* w1T   = (unsigned short*)(ws + 8388608);    // 256 KB  [256][512]
    float*          h_pre = (float*)         (ws + 8650752);    //  8 MB   [8192][256]
    unsigned short* h_bf  = (unsigned short*)(ws + 17039360);   //  4 MB
    unsigned short* w2T   = (unsigned short*)(ws + 21233664);   // 32 KB   [64][256]
    float*          z_pre = (float*)         (ws + 21266432);   //  2 MB   [8192][64]
    unsigned short* z_bf  = (unsigned short*)(ws + 23363584);   //  1 MB
    int*            rptr  = (int*)           (ws + 24412160);   // 8193 ints

    // 1) precision/layout prep
    {
        int n = N_NODES * IN_DIM;
        cvt_f32_bf16<<<(n + 255) / 256, 256, 0, stream>>>(x, x_bf, n);
    }
    transpose_cvt_bf16<<<(IN_DIM * HID_DIM + 255) / 256, 256, 0, stream>>>(w1, w1T, IN_DIM, HID_DIM);
    transpose_cvt_bf16<<<(HID_DIM * Z_DIM + 255) / 256, 256, 0, stream>>>(w2, w2T, HID_DIM, Z_DIM);
    build_row_ptr<<<(N_NODES + 1 + 255) / 256, 256, 0, stream>>>(erow, rptr, N_EDGES, N_NODES);

    // 2) GEMM1: h_pre = x @ W1      (M=8192, N=256, K=512) — h_pre re-read, keep RT stores
    gemm_nt_bf16_wmma<false><<<dim3(HID_DIM / 64, N_NODES / 128), 256, 0, stream>>>(
        x_bf, w1T, h_pre, N_NODES, HID_DIM, IN_DIM, HID_DIM);

    // 3) SpMM1 + ReLU -> h (bf16 only; only consumed by GEMM2)
    spmm_kernel<HID_DIM, true, false><<<N_NODES, HID_DIM, 0, stream>>>(
        h_pre, ecol, ew, rptr, nullptr, h_bf);

    // 4) GEMM2: z_pre = h @ W2      (M=8192, N=64, K=256)
    gemm_nt_bf16_wmma<false><<<dim3(1, N_NODES / 128), 256, 0, stream>>>(
        h_bf, w2T, z_pre, N_NODES, Z_DIM, HID_DIM, Z_DIM);

    // 5) SpMM2 -> z (f32 to d_out tail) + z_bf16 for decoder
    spmm_kernel<Z_DIM, false, true><<<N_NODES, Z_DIM, 0, stream>>>(
        z_pre, ecol, ew, rptr, z_out, z_bf);

    // 6) Decoder: recon = z @ z^T   (M=N=8192, K=64; store-bound ~268 MB)
    //    Non-temporal stores: write-once stream, keep L2 for z/edges/gathers.
    gemm_nt_bf16_wmma<true><<<dim3(N_NODES / 64, N_NODES / 128), 256, 0, stream>>>(
        z_bf, z_bf, recon, N_NODES, N_NODES, Z_DIM, N_NODES);
}
